// VisualizeInternalGatesSentimentClassificaton_60378650247502
// MI455X (gfx1250) — matrix operand order repaired and checked by reading, compiled, not run on hardware
//
#include <hip/hip_runtime.h>
#include <stdint.h>
#include <stddef.h>

#define VOCAB 32000
#define EMB   512
#define HID   768
#define H4    3072      // 4*HID
#define BATCH 64
#define SEQ   256

typedef __attribute__((ext_vector_type(16))) __bf16 v16bf;
typedef __attribute__((ext_vector_type(8)))  float  v8f;

union Frag { uint32_t u[8]; v16bf v; };

__device__ __forceinline__ uint16_t f2bf(float x) {
  uint32_t u = __float_as_uint(x);
  u += 0x7FFFu + ((u >> 16) & 1u);       // round-to-nearest-even
  return (uint16_t)(u >> 16);
}

// A-matrix fragment, 16x32 bf16, row-major source with leading dim lda.
// ISA 7.12.2: lane(0-15)=row m, hi half lanes carry K offset +8;
// VGPR j holds K pair: (j>=4?16:0) + (j&3)*2 + hi*8.
__device__ __forceinline__ v16bf loadA(const uint16_t* __restrict__ A, int lda,
                                       int m0, int k0) {
  int lane = threadIdx.x & 31;
  int m = lane & 15, hi = lane >> 4;
  const uint16_t* p = A + (size_t)(m0 + m) * lda + k0 + hi * 8;
  Frag f;
#pragma unroll
  for (int j = 0; j < 8; ++j) {
    int k = ((j & 4) << 2) + ((j & 3) << 1);   // 0,2,4,6,16,18,20,22
    f.u[j] = *(const uint32_t*)(p + k);
  }
  return f.v;
}

// B-matrix fragment, 32x16 bf16 (KxN), B[k][n] = W[n][k], W row-major ldw.
// ISA: lane = column n (hi half: K offset +8); VGPR j holds K pair: (j>=4?16:0) + (j&3)*2 + hi*8.
__device__ __forceinline__ v16bf loadB(const uint16_t* __restrict__ W, int ldw,
                                       int n0, int k0) {
  int lane = threadIdx.x & 31;
  int n = lane & 15, hi = lane >> 4;
  const uint16_t* p = W + (size_t)(n0 + n) * ldw + k0 + hi * 8;
  Frag f;
#pragma unroll
  for (int j = 0; j < 8; ++j) f.u[j] = *(const uint32_t*)(p + ((j & 4) << 2) + ((j & 3) << 1));
  return f.v;
}

#define WMMA_BF16(acc, a, b) \
  acc = __builtin_amdgcn_wmma_f32_16x16x32_bf16(false, (a), false, (b), (short)0, (acc), false, false)

// ---------------- prep kernels ----------------
__global__ void k_f32_to_bf16(const float* __restrict__ src,
                              uint16_t* __restrict__ dst, int n) {
  int i = blockIdx.x * blockDim.x + threadIdx.x;
  if (i < n) dst[i] = f2bf(src[i]);
}

// Xb[t*64+b][k] = bf16(emb[token[b][t]][k]), padding token 0 -> 0
__global__ void k_gather(const int* __restrict__ tok,
                         const float* __restrict__ emb,
                         uint16_t* __restrict__ Xb) {
  int idx = blockIdx.x * blockDim.x + threadIdx.x;   // over SEQ*BATCH*EMB
  int r = idx >> 9;          // EMB = 512
  int k = idx & 511;
  int t = r >> 6, b = r & 63;
  int tk = tok[b * SEQ + t];
  float v = (tk == 0) ? 0.f : emb[(size_t)tk * EMB + k];
  Xb[(size_t)r * EMB + k] = f2bf(v);
}

__global__ void k_init(uint16_t* __restrict__ hbf, float* __restrict__ c) {
  int i = blockIdx.x * blockDim.x + threadIdx.x;
  if (i < BATCH * HID) { hbf[i] = 0; c[i] = 0.f; }
}

// ---------------- phase 1: Zx = Xb @ W_ih^T + (b_ih + b_hh) ----------------
// grid (256, 96), block 256 (8 waves, 4 M x 2 N tiles); K = 512
__global__ void k_gemm_zx(const uint16_t* __restrict__ Xb,
                          const uint16_t* __restrict__ Wih,
                          const float* __restrict__ bih,
                          const float* __restrict__ bhh,
                          float* __restrict__ Zx) {
  int wave = threadIdx.x >> 5;
  int m0 = blockIdx.x * 64 + (wave & 3) * 16;
  int n0 = blockIdx.y * 32 + (wave >> 2) * 16;
  v8f acc = {};
  for (int k0 = 0; k0 < EMB; k0 += 32) {
    v16bf a = loadA(Xb, EMB, m0, k0);
    v16bf b = loadB(Wih, EMB, n0, k0);
    WMMA_BF16(acc, a, b);
  }
  int lane = threadIdx.x & 31;
  int n = n0 + (lane & 15), hi = lane >> 4;
  float bias = bih[n] + bhh[n];
#pragma unroll
  for (int j = 0; j < 8; ++j)
    Zx[(size_t)(m0 + hi * 8 + j) * H4 + n] = acc[j] + bias;
}

// ---------------- recurrent step ----------------
// grid = 12 (h-chunks of 64 cols), block 256 (8 waves).
// z = Zx[t] + h_prev @ W_hh^T for 4 gate column groups; exchange via LDS;
// then elementwise gates + state update + gate-stream stores.
__global__ void k_step(const float* __restrict__ Zx_t,
                       const uint16_t* __restrict__ Whh,
                       const uint16_t* __restrict__ h_prev,
                       uint16_t* __restrict__ h_next,
                       float* __restrict__ c,
                       float* __restrict__ h_f32,
                       float* __restrict__ out_g, float* __restrict__ out_f,
                       float* __restrict__ out_i, float* __restrict__ out_o) {
  __shared__ float zsh[64 * 256];        // [batch 64][gate*64 + hcol] = 64 KB
  const int hblk = blockIdx.x;           // h columns hblk*64 .. +63
  const int wave = threadIdx.x >> 5;
  const int lane = threadIdx.x & 31;

  for (int tile = wave; tile < 64; tile += 8) {
    int gate = tile >> 4;                // 0..3 (i,f,g,o column groups)
    int mt   = (tile >> 2) & 3;          // batch tile
    int nt   = tile & 3;                 // 16-col tile inside the 64-col chunk
    int m0 = mt * 16;
    int n0 = gate * HID + hblk * 64 + nt * 16;   // row in W_hh / col in z
    v8f acc = {};
    for (int k0 = 0; k0 < HID; k0 += 32) {
      v16bf a = loadA(h_prev, HID, m0, k0);
      v16bf b = loadB(Whh, HID, n0, k0);
      WMMA_BF16(acc, a, b);
    }
    int n = lane & 15, hi = lane >> 4;
    int col = gate * 64 + nt * 16 + n;
#pragma unroll
    for (int j = 0; j < 8; ++j) {
      int m = m0 + hi * 8 + j;
      zsh[m * 256 + col] = acc[j] + Zx_t[(size_t)m * H4 + n0 + n];
    }
  }
  __syncthreads();

  for (int idx = threadIdx.x; idx < 64 * 64; idx += 256) {
    int b = idx >> 6, hc = idx & 63;
    int h = hblk * 64 + hc;
    float zi = zsh[b * 256 + hc];
    float zf = zsh[b * 256 + 64  + hc];
    float zg = zsh[b * 256 + 128 + hc];
    float zo = zsh[b * 256 + 192 + hc];
    float ig = 1.f / (1.f + __expf(-zi));
    float fg = 1.f / (1.f + __expf(-zf));
    float gg = tanhf(zg);
    float og = 1.f / (1.f + __expf(-zo));
    size_t ci = (size_t)b * HID + h;
    float cn = fg * c[ci] + ig * gg;
    float hn = og * tanhf(cn);
    c[ci]     = cn;
    h_f32[ci] = hn;
    h_next[ci] = f2bf(hn);
    out_g[ci] = gg; out_f[ci] = fg; out_i[ci] = ig; out_o[ci] = og;
  }
}

// ---------------- logits ----------------
__global__ void k_logits(const float* __restrict__ h,
                         const float* __restrict__ Wcls,
                         const float* __restrict__ bcls,
                         float* __restrict__ out) {
  int tid = threadIdx.x;                 // 128 threads: (b, class)
  if (tid >= 128) return;
  int b = tid >> 1, cls = tid & 1;
  float s = bcls[cls];
  for (int k = 0; k < HID; ++k)
    s += h[(size_t)b * HID + k] * Wcls[(size_t)cls * HID + k];
  out[b * 2 + cls] = s;
}

extern "C" void kernel_launch(void* const* d_in, const int* in_sizes, int n_in,
                              void* d_out, int out_size, void* d_ws, size_t ws_size,
                              hipStream_t stream) {
  (void)in_sizes; (void)n_in; (void)out_size; (void)ws_size;

  const int*   tok  = (const int*)  d_in[0];
  const float* emb  = (const float*)d_in[1];
  const float* Wih  = (const float*)d_in[2];
  const float* Whh  = (const float*)d_in[3];
  const float* bih  = (const float*)d_in[4];
  const float* bhh  = (const float*)d_in[5];
  const float* Wcls = (const float*)d_in[6];
  const float* bcls = (const float*)d_in[7];

  char* ws = (char*)d_ws;
  size_t off = 0;
  auto alloc = [&](size_t bytes) -> char* {
    char* p = ws + off;
    off += (bytes + 255) & ~(size_t)255;
    return p;
  };

  uint16_t* Wih_bf = (uint16_t*)alloc((size_t)H4 * EMB * 2);        //  3.1 MB
  uint16_t* Whh_bf = (uint16_t*)alloc((size_t)H4 * HID * 2);        //  4.7 MB
  uint16_t* Xb     = (uint16_t*)alloc((size_t)SEQ * BATCH * EMB * 2); // 16.8 MB
  float*    Zx     = (float*)   alloc((size_t)SEQ * BATCH * H4 * 4);  // 201 MB
  uint16_t* hbf    = (uint16_t*)alloc((size_t)2 * BATCH * HID * 2); // double-buffered
  float*    cbuf   = (float*)   alloc((size_t)BATCH * HID * 4);
  float*    hf     = (float*)   alloc((size_t)BATCH * HID * 4);

  // prep
  k_f32_to_bf16<<<(H4 * EMB + 255) / 256, 256, 0, stream>>>(Wih, Wih_bf, H4 * EMB);
  k_f32_to_bf16<<<(H4 * HID + 255) / 256, 256, 0, stream>>>(Whh, Whh_bf, H4 * HID);
  k_gather<<<(SEQ * BATCH * EMB) / 256, 256, 0, stream>>>(tok, emb, Xb);
  k_init<<<(BATCH * HID + 255) / 256, 256, 0, stream>>>(hbf, cbuf);

  // hoisted input projection: Zx = Xb @ W_ih^T + (b_ih + b_hh)
  k_gemm_zx<<<dim3(SEQ * BATCH / 64, H4 / 32), 256, 0, stream>>>(Xb, Wih_bf, bih, bhh, Zx);

  // recurrence
  float* out = (float*)d_out;
  float* og  = out + 128;                        // gate streams after logits
  const size_t GSZ = (size_t)SEQ * BATCH * HID;  // one [T,B,H] stream
  for (int t = 0; t < SEQ; ++t) {
    const uint16_t* hp = hbf + (size_t)(t & 1) * BATCH * HID;
    uint16_t*       hn = hbf + (size_t)((t + 1) & 1) * BATCH * HID;
    size_t toff = (size_t)t * BATCH * HID;
    k_step<<<HID / 64, 256, 0, stream>>>(
        Zx + (size_t)t * BATCH * H4, Whh_bf, hp, hn, cbuf, hf,
        og + toff, og + GSZ + toff, og + 2 * GSZ + toff, og + 3 * GSZ + toff);
  }

  k_logits<<<1, 128, 0, stream>>>(hf, Wcls, bcls, out);
}
